// Seq2SeqPointCE_1288490189164
// MI455X (gfx1250) — compile-verified, matrix operand order checked by reading, not run on hardware
//
#include <hip/hip_runtime.h>
#include <hip/hip_bf16.h>

// Problem dims (fixed by reference)
#define VOCAB 32000
#define EE 256
#define HH 256
#define BB 64
#define SS 512
#define TT 256
#define N3H 768          // 3*H
#define KT  8            // 256 / 32 K-steps per WMMA pass

typedef __attribute__((ext_vector_type(16))) _Float16 v16h;
typedef __attribute__((ext_vector_type(8)))  _Float16 v8h;
typedef __attribute__((ext_vector_type(8)))  float    v8f;

// ---------------------------------------------------------------------------
// Weight packing: W (N3H x K) row-major fp32  ->  per-lane WMMA B-operand f16
// tile (ct, kt): 32 lanes x 16 halfs contiguous.
//   lane l (<16): N = ct*16 + l,     K-locals = {0..7, 16..23}
//   lane l (>=16): N = ct*16 + l-16, K-locals = {8..15, 24..31}
// j-th half: j<8 -> k = kbase + j ; j>=8 -> k = kbase + 16 + (j-8)
// ---------------------------------------------------------------------------
__global__ void pack_weights(const float* __restrict__ W, _Float16* __restrict__ pk,
                             int N, int K) {
  int idx = blockIdx.x * blockDim.x + threadIdx.x;
  int total = N * K;
  if (idx >= total) return;
  int j    = idx & 15;
  int lane = (idx >> 4) & 31;
  int tile = idx >> 9;            // ct*KT + kt
  int kt   = tile & (KT - 1);
  int ct   = tile >> 3;
  int nrow = ct * 16 + (lane & 15);
  int kbase = 8 * (lane >> 4);
  int k = kt * 32 + ((j < 8) ? (kbase + j) : (kbase + 16 + (j - 8)));
  pk[idx] = (_Float16)W[nrow * K + k];
}

// A-operand gather from LDS row-major f16 buffer (rows of length HH).
__device__ __forceinline__ v16h load_a(const _Float16* row_base, int off) {
  v8h lo = *(const v8h*)(row_base + off);
  v8h hi = *(const v8h*)(row_base + off + 16);
  v16h r;
#pragma unroll
  for (int i = 0; i < 8; ++i) { r[i] = lo[i]; r[i + 8] = hi[i]; }
  return r;
}

__device__ __forceinline__ float sigm(float x) { return 1.f / (1.f + __expf(-x)); }

struct GruDirParams {
  const int*      toks;     // encoder: (B,S) seq ; decoder: trg (B,T)
  const _Float16* pkWi;     // packed Wi^T, f16
  const _Float16* pkWh;     // packed Wh^T, f16
  const float*    bi;       // (3H)
  const float*    bh;       // (3H)
  float*          score;    // (B,L): encoder A-slice, decoder s
  float*          hfinal;   // (B,H) or nullptr
  const float*    h_init;   // decoder h0 (B,H) or nullptr
};

// ---------------------------------------------------------------------------
// Persistent per-16-row GRU recurrence. One block = one WMMA M-tile (16 batch
// rows), 16 waves; wave w owns gate-column tile ct=w for all 3 gates, so the
// full r/z/n fusion happens in registers with zero cross-wave traffic.
// x_{t+1} is software-pipelined through registers across the WMMA phase.
// ---------------------------------------------------------------------------
__global__ __launch_bounds__(512)
void gru_kernel(GruDirParams pa, GruDirParams pb,
                const float* __restrict__ emb,
                const float* __restrict__ whid,     // H floats (enc or dec half)
                const int*   __restrict__ pre_seq,  // for decoder t==0 token
                int L, int mode) {                  // mode: 0 encoder, 1 decoder
  __shared__ __align__(16) _Float16 xbuf[16 * HH];
  __shared__ __align__(16) _Float16 hbuf[16 * HH];
  __shared__ __align__(16) float    hf[16 * HH];
  __shared__ float wh_s[HH];

  GruDirParams P = (blockIdx.x < 4) ? pa : pb;
  const int rbase = (blockIdx.x & 3) * 16;
  const int tid  = threadIdx.x;
  const int lane = tid & 31;
  const int w    = tid >> 5;          // wave id == column tile ct (0..15)

  for (int i = tid; i < HH; i += 512) wh_s[i] = whid[i];
  for (int i = tid; i < 16 * HH; i += 512) {
    float hv = (mode == 1) ? P.h_init[(rbase + i / HH) * HH + (i % HH)] : 0.f;
    hf[i] = hv; hbuf[i] = (_Float16)hv;
  }

  // x-gather assignment: wave w stages row w; lane covers 8 consecutive cols.
  const int grow = w;                  // 0..15
  const int gcol = lane * 8;           // 0..248
  const int gb   = rbase + grow;

  const int arow  = lane & 15;
  const int kbsel = (lane >> 4) * 8;
  const int c     = w * 16 + (lane & 15);
  const float bir = P.bi[c],        bhr = P.bh[c];
  const float biz = P.bi[256 + c],  bhz = P.bh[256 + c];
  const float bin = P.bi[512 + c],  bhn = P.bh[512 + c];

  // prologue: stage x_0 in registers
  float4 xa, xb;
  {
    int tok = (mode == 1) ? pre_seq[gb * SS + (SS - 1)] : P.toks[gb * L];
    const float4* e = (const float4*)(emb + (long)tok * EE + gcol);
    xa = e[0]; xb = e[1];
  }
  __syncthreads();

  for (int t = 0; t < L; ++t) {
    // --- commit staged x_t to LDS as f16 (one ds_store_b128 per thread) ---
    {
      v8h hx;
      hx[0] = (_Float16)xa.x; hx[1] = (_Float16)xa.y;
      hx[2] = (_Float16)xa.z; hx[3] = (_Float16)xa.w;
      hx[4] = (_Float16)xb.x; hx[5] = (_Float16)xb.y;
      hx[6] = (_Float16)xb.z; hx[7] = (_Float16)xb.w;
      *(v8h*)(xbuf + grow * HH + gcol) = hx;
    }
    __syncthreads();

    // --- issue x_{t+1} loads now; consumed next iteration (latency hidden) -
    if (t + 1 < L) {
      int tok = (mode == 1) ? P.toks[gb * L + t] : P.toks[gb * L + t + 1];
      const float4* e = (const float4*)(emb + (long)tok * EE + gcol);
      xa = e[0]; xb = e[1];
    }

    // --- WMMA: 6 accumulators (gi,gh) x (r,z,n), K = 256 -------------------
    v8f zero = {0.f, 0.f, 0.f, 0.f, 0.f, 0.f, 0.f, 0.f};
    v8f aI0 = zero, aI1 = zero, aI2 = zero;
    v8f aH0 = zero, aH1 = zero, aH2 = zero;
    const _Float16* xr = xbuf + arow * HH;
    const _Float16* hr = hbuf + arow * HH;
#pragma unroll
    for (int kt = 0; kt < KT; ++kt) {
      const int koff = kt * 32 + kbsel;
      v16h ax = load_a(xr, koff);
      v16h ah = load_a(hr, koff);
#pragma unroll
      for (int g = 0; g < 3; ++g) {
        const int ctg = g * 16 + w;
        const long boff = ((long)(ctg * KT + kt) * 32 + lane) * 16;
        v16h bwi = *(const v16h*)(P.pkWi + boff);
        v16h bwh = *(const v16h*)(P.pkWh + boff);
        if (g == 0) {
          aI0 = __builtin_amdgcn_wmma_f32_16x16x32_f16(false, ax, false, bwi, (short)0, aI0, false, false);
          aH0 = __builtin_amdgcn_wmma_f32_16x16x32_f16(false, ah, false, bwh, (short)0, aH0, false, false);
        } else if (g == 1) {
          aI1 = __builtin_amdgcn_wmma_f32_16x16x32_f16(false, ax, false, bwi, (short)0, aI1, false, false);
          aH1 = __builtin_amdgcn_wmma_f32_16x16x32_f16(false, ah, false, bwh, (short)0, aH1, false, false);
        } else {
          aI2 = __builtin_amdgcn_wmma_f32_16x16x32_f16(false, ax, false, bwi, (short)0, aI2, false, false);
          aH2 = __builtin_amdgcn_wmma_f32_16x16x32_f16(false, ah, false, bwh, (short)0, aH2, false, false);
        }
      }
    }
    __syncthreads();   // all hbuf reads done before we overwrite h

    // --- fused gate math; lane owns column c, rows v + 8*(lane>>4) ---------
#pragma unroll
    for (int v = 0; v < 8; ++v) {
      int m = v + 8 * (lane >> 4);
      float rg = sigm(aI0[v] + bir + aH0[v] + bhr);
      float zg = sigm(aI1[v] + biz + aH1[v] + bhz);
      float ng = tanhf(aI2[v] + bin + rg * (aH2[v] + bhn));
      float hold = hf[m * HH + c];
      float hnew = (1.f - zg) * ng + zg * hold;
      hf[m * HH + c]   = hnew;
      hbuf[m * HH + c] = (_Float16)hnew;
    }

    // --- L2 lookahead for x_{t+2} embedding row (off the critical path) ----
    if (t + 2 < L) {
      int tok2 = (mode == 1) ? P.toks[gb * L + t + 1] : P.toks[gb * L + t + 2];
      __builtin_prefetch(emb + (long)tok2 * EE + gcol, 0, 0);
    }
    __syncthreads();

    // --- score: wave r reduces row r with shuffle butterfly ----------------
    {
      float partial = 0.f;
#pragma unroll
      for (int c8 = 0; c8 < HH / 32; ++c8) {
        int cc = lane + c8 * 32;
        float hv = hf[w * HH + cc];
        partial += (hv > 0.f ? hv : 0.f) * wh_s[cc];
      }
#pragma unroll
      for (int o = 16; o > 0; o >>= 1) partial += __shfl_down(partial, o, 32);
      if (lane == 0) P.score[(rbase + w) * L + t] = partial;
    }
    __syncthreads();
  }

  if (P.hfinal)
    for (int i = tid; i < 16 * HH; i += 512)
      P.hfinal[(rbase + i / HH) * HH + (i % HH)] = hf[i];
}

// ---------------------------------------------------------------------------
// h0 = tanh(cat(pre_h, post_h) @ Wfc^T + bfc)   (64x256, K=512 — tiny)
// ---------------------------------------------------------------------------
__global__ void h0_kernel(const float* __restrict__ hpre, const float* __restrict__ hpost,
                          const float* __restrict__ Wfc, const float* __restrict__ bfc,
                          float* __restrict__ h0) {
  int idx = blockIdx.x * blockDim.x + threadIdx.x;
  if (idx >= BB * HH) return;
  int b = idx >> 8, j = idx & 255;
  const float* wr = Wfc + j * (2 * HH);
  float s = bfc[j];
  for (int k = 0; k < HH; ++k) s += hpre[b * HH + k] * wr[k];
  for (int k = 0; k < HH; ++k) s += hpost[b * HH + k] * wr[HH + k];
  h0[idx] = tanhf(s);
}

// ---------------------------------------------------------------------------
// out[b,t,j]  = (A_pre [b,j] + s[b,t] + b_hid) * w_out + b_out   (float4)
// out2[b,t,j] = (A_post[b,j] + s[b,t] + b_hid) * w_out + b_out
// ---------------------------------------------------------------------------
__global__ void out_kernel(const float4* __restrict__ Apre, const float4* __restrict__ Apost,
                           const float* __restrict__ sv,
                           const float* __restrict__ bhid, const float* __restrict__ wout,
                           const float* __restrict__ bout, float4* __restrict__ out) {
  const long total4 = (long)BB * TT * (SS / 4);
  const int  S4 = SS / 4;
  const float bh = bhid[0], wo = wout[0], bo = bout[0];
  for (long i = blockIdx.x * (long)blockDim.x + threadIdx.x; i < total4;
       i += (long)gridDim.x * blockDim.x) {
    long j4 = i % S4; long bt = i / S4; long t = bt % TT; long b = bt / TT;
    float base = (sv[b * TT + t] + bh) * wo + bo;   // (A + base')*wo+bo = A*wo + base
    float4 ap = Apre [b * S4 + j4];
    float4 aq = Apost[b * S4 + j4];
    float4 o0 = { ap.x * wo + base, ap.y * wo + base, ap.z * wo + base, ap.w * wo + base };
    float4 o1 = { aq.x * wo + base, aq.y * wo + base, aq.z * wo + base, aq.w * wo + base };
    out[i]          = o0;
    out[total4 + i] = o1;
  }
}

// ---------------------------------------------------------------------------
extern "C" void kernel_launch(void* const* d_in, const int* in_sizes, int n_in,
                              void* d_out, int out_size, void* d_ws, size_t ws_size,
                              hipStream_t stream) {
  const int*   pre_seq  = (const int*)  d_in[0];
  const int*   post_seq = (const int*)  d_in[1];
  const int*   trg      = (const int*)  d_in[2];
  const float* emb      = (const float*)d_in[3];
  const float* Wi_pre   = (const float*)d_in[4];
  const float* Wh_pre   = (const float*)d_in[5];
  const float* bi_pre   = (const float*)d_in[6];
  const float* bh_pre   = (const float*)d_in[7];
  const float* Wi_post  = (const float*)d_in[8];
  const float* Wh_post  = (const float*)d_in[9];
  const float* bi_post  = (const float*)d_in[10];
  const float* bh_post  = (const float*)d_in[11];
  const float* Wfc      = (const float*)d_in[12];
  const float* bfc      = (const float*)d_in[13];
  const float* Wi_dec   = (const float*)d_in[14];
  const float* Wh_dec   = (const float*)d_in[15];
  const float* bi_dec   = (const float*)d_in[16];
  const float* bh_dec   = (const float*)d_in[17];
  const float* w_hid    = (const float*)d_in[18];
  const float* b_hid    = (const float*)d_in[19];
  const float* w_out    = (const float*)d_in[20];
  const float* b_out    = (const float*)d_in[21];
  float* out = (float*)d_out;

  // workspace carve-up
  char* ws = (char*)d_ws;
  const size_t PK = (size_t)N3H * HH * sizeof(_Float16);   // 384 KB per matrix
  _Float16* pkWi_pre  = (_Float16*)(ws);            _Float16* pkWh_pre  = (_Float16*)(ws + 1*PK);
  _Float16* pkWi_post = (_Float16*)(ws + 2*PK);     _Float16* pkWh_post = (_Float16*)(ws + 3*PK);
  _Float16* pkWi_dec  = (_Float16*)(ws + 4*PK);     _Float16* pkWh_dec  = (_Float16*)(ws + 5*PK);
  char* p = ws + 6*PK;
  float* A_pre  = (float*)p;              p += (size_t)BB*SS*sizeof(float);
  float* A_post = (float*)p;              p += (size_t)BB*SS*sizeof(float);
  float* h_pre  = (float*)p;              p += (size_t)BB*HH*sizeof(float);
  float* h_post = (float*)p;              p += (size_t)BB*HH*sizeof(float);
  float* h0     = (float*)p;              p += (size_t)BB*HH*sizeof(float);
  float* s_dec  = (float*)p;              p += (size_t)BB*TT*sizeof(float);

  // 1) pack weights into WMMA B-operand layout (f16)
  {
    int total = N3H * HH, blk = 256, grd = (total + blk - 1) / blk;
    pack_weights<<<grd, blk, 0, stream>>>(Wi_pre,  pkWi_pre,  N3H, HH);
    pack_weights<<<grd, blk, 0, stream>>>(Wh_pre,  pkWh_pre,  N3H, HH);
    pack_weights<<<grd, blk, 0, stream>>>(Wi_post, pkWi_post, N3H, HH);
    pack_weights<<<grd, blk, 0, stream>>>(Wh_post, pkWh_post, N3H, HH);
    pack_weights<<<grd, blk, 0, stream>>>(Wi_dec,  pkWi_dec,  N3H, HH);
    pack_weights<<<grd, blk, 0, stream>>>(Wh_dec,  pkWh_dec,  N3H, HH);
  }

  // 2) both encoders concurrently: grid = 8 (pre: blocks 0-3, post: 4-7)
  {
    GruDirParams pa = { pre_seq,  pkWi_pre,  pkWh_pre,  bi_pre,  bh_pre,  A_pre,  h_pre,  nullptr };
    GruDirParams pb = { post_seq, pkWi_post, pkWh_post, bi_post, bh_post, A_post, h_post, nullptr };
    gru_kernel<<<8, 512, 0, stream>>>(pa, pb, emb, w_hid, pre_seq, SS, 0);
  }

  // 3) decoder initial hidden
  h0_kernel<<<(BB * HH + 255) / 256, 256, 0, stream>>>(h_pre, h_post, Wfc, bfc, h0);

  // 4) decoder: grid = 4
  {
    GruDirParams pa = { trg, pkWi_dec, pkWh_dec, bi_dec, bh_dec, s_dec, nullptr, h0 };
    gru_kernel<<<4, 512, 0, stream>>>(pa, pa, emb, w_hid + HH, pre_seq, TT, 1);
  }

  // 5) broadcast output (bandwidth bound, ~67 MB at 23.3 TB/s)
  out_kernel<<<8192, 256, 0, stream>>>((const float4*)A_pre, (const float4*)A_post,
                                       s_dec, b_hid, w_out, b_out, (float4*)out);
}
